// GAT_3968549782307
// MI455X (gfx1250) — matrix operand order repaired and checked by reading, not compiled, not run on hardware
//
#include <hip/hip_runtime.h>

// GAT layer-0 forward for MI455X (gfx1250, wave32).
// out = elu(segment_softmax_aggregate(X @ W0^T)) ; layer 1 of the reference is
// discarded by the reference, so it is not computed.
//
// Pipeline (all on `stream`, d_ws scratch):
//   K0 split f32 -> (bf16 hi, bf16 lo) for X and W0 (one-shot)
//   K1 gemm: H = X @ W0^T via v_wmma_f32_16x16x32_bf16, 3 products
//            (Xh*Wh + Xh*Wl + Xl*Wh) ~ f32 accuracy, K fully unrolled
//   K2 attention dots a_s, a_d                                   [N,8]
//   K3 init: m=enc(-inf), den=0, out=0
//   K4 edge pass 1: atomicMax segment max of leaky_relu logits
//   K5 edge pass 2: atomicAdd segment sum of exp(logit - m)
//   K6 edge pass 3: wave-per-(edge,head) gather/scatter out += alpha*H[src]
//   K7 out = elu(out + b0)

#define D_INK   256   // input dim = K
#define D_HIDK  256   // HEADS*HID = output cols
#define HEADSK  8
#define HIDK    32
#define NEG_SLOPE 0.2f

typedef __attribute__((ext_vector_type(16))) __bf16          v16bf;
typedef __attribute__((ext_vector_type(16))) unsigned short  v16us;
typedef __attribute__((ext_vector_type(8)))  unsigned short  v8us;
typedef __attribute__((ext_vector_type(8)))  float           v8f;

__device__ __forceinline__ unsigned short f2bf(float f) {
  unsigned u = __float_as_uint(f);
  unsigned r = 0x7fffu + ((u >> 16) & 1u);          // round-to-nearest-even
  return (unsigned short)((u + r) >> 16);
}
__device__ __forceinline__ float bf2f(unsigned short b) {
  return __uint_as_float(((unsigned)b) << 16);
}
// monotone f32 <-> u32 encoding for atomicMax on floats
__device__ __forceinline__ unsigned enc_f(float f) {
  unsigned u = __float_as_uint(f);
  return (u & 0x80000000u) ? ~u : (u | 0x80000000u);
}
__device__ __forceinline__ float dec_f(unsigned k) {
  return (k & 0x80000000u) ? __uint_as_float(k & 0x7fffffffu)
                           : __uint_as_float(~k);
}
#define ENC_NEG_INF 0x007fffffu   // enc_f(-inf)

__device__ __forceinline__ float lrelu(float x) {
  return x > 0.0f ? x : NEG_SLOPE * x;
}

// ----------------------------------------------- K0: f32 -> bf16 hi/lo split
__global__ void gat0_split_bf16(const float* __restrict__ src,
                                unsigned short* __restrict__ hi,
                                unsigned short* __restrict__ lo, int n) {
  int i = blockIdx.x * blockDim.x + threadIdx.x;
  if (i >= n) return;
  float f = src[i];
  unsigned short h = f2bf(f);
  hi[i] = h;
  lo[i] = f2bf(f - bf2f(h));
}

// ---------------------------------------------------------------- K1: GEMM
// One wave computes one 16x16 tile of H = X @ W0^T from pre-split bf16.
__global__ __launch_bounds__(256)
void gat0_gemm_wmma(const unsigned short* __restrict__ Xh,
                    const unsigned short* __restrict__ Xl,
                    const unsigned short* __restrict__ Wh,
                    const unsigned short* __restrict__ Wl,
                    float* __restrict__ H, int N) {
  int wid = blockIdx.x * 8 + (threadIdx.x >> 5);
  int totalTiles = (N >> 4) * 16;                 // 16 col-tiles (256/16)
  if (wid >= totalTiles) return;
  int tile_m = wid >> 4;
  int tile_o = wid & 15;

  int lane = threadIdx.x & 31;
  int half = lane >> 4;                           // 0 | 1
  int idx  = lane & 15;                           // row (A) / col (B)

  const unsigned short* xh = Xh + (size_t)(tile_m * 16 + idx) * D_INK;
  const unsigned short* xl = Xl + (size_t)(tile_m * 16 + idx) * D_INK;
  const unsigned short* wh = Wh + (size_t)(tile_o * 16 + idx) * D_INK;
  const unsigned short* wl = Wl + (size_t)(tile_o * 16 + idx) * D_INK;

  v8f c = {};
#pragma unroll
  for (int k0 = 0; k0 < D_INK; k0 += 32) {
    // A layout: elem i -> K = 8*half + i (i<8) ; 16 + 8*half + (i-8) (i>=8)
    union { v16us v; v8us h[2]; } ahu, alu;
    ahu.h[0] = *(const v8us*)(xh + k0 + 8 * half);
    ahu.h[1] = *(const v8us*)(xh + k0 + 16 + 8 * half);
    alu.h[0] = *(const v8us*)(xl + k0 + 8 * half);
    alu.h[1] = *(const v8us*)(xl + k0 + 16 + 8 * half);
    // B layout: elem i -> K = 8*half + i (i<8) ; 16 + 8*half + (i-8) (i>=8)  (column = idx = one W0 row)
    union { v16us v; v8us h[2]; } bhuu, bluu;
    bhuu.h[0] = *(const v8us*)(wh + k0 + 8 * half);
    bhuu.h[1] = *(const v8us*)(wh + k0 + 16 + 8 * half);
    bluu.h[0] = *(const v8us*)(wl + k0 + 8 * half);
    bluu.h[1] = *(const v8us*)(wl + k0 + 16 + 8 * half);
    v16us bhu = bhuu.v;
    v16us blu = bluu.v;

    v16bf ah = __builtin_bit_cast(v16bf, ahu.v);
    v16bf al = __builtin_bit_cast(v16bf, alu.v);
    v16bf bh = __builtin_bit_cast(v16bf, bhu);
    v16bf bl = __builtin_bit_cast(v16bf, blu);

    c = __builtin_amdgcn_wmma_f32_16x16x32_bf16(false, ah, false, bh,
                                                (short)0, c, false, false);
    c = __builtin_amdgcn_wmma_f32_16x16x32_bf16(false, ah, false, bl,
                                                (short)0, c, false, false);
    c = __builtin_amdgcn_wmma_f32_16x16x32_bf16(false, al, false, bh,
                                                (short)0, c, false, false);
  }
  // C/D: VGPR r -> row 8*half + r, col idx
  float* hout =
      H + (size_t)(tile_m * 16 + 8 * half) * D_HIDK + tile_o * 16 + idx;
#pragma unroll
  for (int r = 0; r < 8; ++r) hout[(size_t)r * D_HIDK] = c[r];
}

// ---------------------------------------------------- K2: attention dots
__global__ void gat0_attn_dot(const float* __restrict__ H,
                              const float* __restrict__ att_s,
                              const float* __restrict__ att_d,
                              float* __restrict__ a_s, float* __restrict__ a_d,
                              int N) {
  int t = blockIdx.x * blockDim.x + threadIdx.x;     // node*8 + head
  if (t >= N * HEADSK) return;
  int h = t & 7;
  int n = t >> 3;
  const float* hp = H + (size_t)n * D_HIDK + h * HIDK;
  const float* sp = att_s + h * HIDK;
  const float* dp = att_d + h * HIDK;
  float ss = 0.0f, sd = 0.0f;
#pragma unroll
  for (int f = 0; f < HIDK; ++f) {
    ss += hp[f] * sp[f];
    sd += hp[f] * dp[f];
  }
  a_s[t] = ss;
  a_d[t] = sd;
}

// ------------------------------------------------------------- K3: init
__global__ void gat0_init(unsigned* __restrict__ m, float* __restrict__ den,
                          float* __restrict__ out, int N) {
  int i = blockIdx.x * blockDim.x + threadIdx.x;
  int ntot = N * D_HIDK;
  if (i >= ntot) return;
  out[i] = 0.0f;
  if (i < N * HEADSK) {
    m[i] = ENC_NEG_INF;
    den[i] = 0.0f;
  }
}

__device__ __forceinline__ void edge_nodes(const int* __restrict__ ei, int e,
                                           int E, int& src, int& dst) {
  if (e < E) {
    src = ei[e];
    dst = ei[E + e];
  } else {            // self loop appended after concat
    src = dst = e - E;
  }
}

// ---------------------------------------------------- K4: segment max
__global__ void gat0_edge_max(const int* __restrict__ ei,
                              const float* __restrict__ a_s,
                              const float* __restrict__ a_d,
                              unsigned* __restrict__ m, int E, int Etot) {
  long long t = (long long)blockIdx.x * blockDim.x + threadIdx.x;
  if (t >= (long long)Etot * HEADSK) return;
  int h = (int)(t & 7);
  int e = (int)(t >> 3);
  int src, dst;
  edge_nodes(ei, e, E, src, dst);
  float lg = lrelu(a_s[src * HEADSK + h] + a_d[dst * HEADSK + h]);
  atomicMax(&m[dst * HEADSK + h], enc_f(lg));
}

// ---------------------------------------------------- K5: segment sum
__global__ void gat0_edge_sum(const int* __restrict__ ei,
                              const float* __restrict__ a_s,
                              const float* __restrict__ a_d,
                              const unsigned* __restrict__ m,
                              float* __restrict__ den, int E, int Etot) {
  long long t = (long long)blockIdx.x * blockDim.x + threadIdx.x;
  if (t >= (long long)Etot * HEADSK) return;
  int h = (int)(t & 7);
  int e = (int)(t >> 3);
  int src, dst;
  edge_nodes(ei, e, E, src, dst);
  float lg = lrelu(a_s[src * HEADSK + h] + a_d[dst * HEADSK + h]);
  float ex = expf(lg - dec_f(m[dst * HEADSK + h]));
  atomicAdd(&den[dst * HEADSK + h], ex);
}

// ------------------------------------- K6: aggregation, wave per (edge,head)
__global__ __launch_bounds__(256)
void gat0_edge_aggr(const int* __restrict__ ei, const float* __restrict__ a_s,
                    const float* __restrict__ a_d,
                    const unsigned* __restrict__ m,
                    const float* __restrict__ den, const float* __restrict__ H,
                    float* __restrict__ out, int E, int Etot) {
  long long wid = (long long)blockIdx.x * 8 + (threadIdx.x >> 5);
  if (wid >= (long long)Etot * HEADSK) return;
  int lane = threadIdx.x & 31;                       // feature index (HID==32)
  int h = (int)(wid & 7);
  int e = (int)(wid >> 3);
  int src, dst;
  edge_nodes(ei, e, E, src, dst);
  float lg = lrelu(a_s[src * HEADSK + h] + a_d[dst * HEADSK + h]);
  float ex = expf(lg - dec_f(m[dst * HEADSK + h]));
  float alpha = ex / (den[dst * HEADSK + h] + 1e-16f);
  float v = alpha * H[(size_t)src * D_HIDK + h * HIDK + lane];
  atomicAdd(&out[(size_t)dst * D_HIDK + h * HIDK + lane], v);
}

// ------------------------------------------------------------ K7: elu+bias
__global__ void gat0_elu_bias(float* __restrict__ out,
                              const float* __restrict__ b0, int ntot) {
  int i = blockIdx.x * blockDim.x + threadIdx.x;
  if (i >= ntot) return;
  float v = out[i] + b0[i & (D_HIDK - 1)];
  out[i] = v > 0.0f ? v : expm1f(v);
}

extern "C" void kernel_launch(void* const* d_in, const int* in_sizes, int n_in,
                              void* d_out, int out_size, void* d_ws,
                              size_t ws_size, hipStream_t stream) {
  const float* x   = (const float*)d_in[0];
  const int*   ei  = (const int*)d_in[1];
  const float* W0  = (const float*)d_in[2];
  const float* as0 = (const float*)d_in[3];
  const float* ad0 = (const float*)d_in[4];
  const float* b0  = (const float*)d_in[5];
  float* out = (float*)d_out;

  int N = in_sizes[0] / D_INK;       // 50000
  int E = in_sizes[1] / 2;           // 800000
  int Etot = E + N;                  // with self loops

  // workspace layout (all slices 256B-aligned by construction)
  char* ws = (char*)d_ws;
  size_t off = 0;
  auto alloc = [&](size_t bytes) {
    void* p = ws + off;
    off += (bytes + 255) & ~(size_t)255;
    return p;
  };
  float*          H    = (float*)alloc((size_t)N * D_HIDK * 4);
  float*          a_s  = (float*)alloc((size_t)N * HEADSK * 4);
  float*          a_d  = (float*)alloc((size_t)N * HEADSK * 4);
  unsigned*       mseg = (unsigned*)alloc((size_t)N * HEADSK * 4);
  float*          den  = (float*)alloc((size_t)N * HEADSK * 4);
  unsigned short* Xh   = (unsigned short*)alloc((size_t)N * D_INK * 2);
  unsigned short* Xl   = (unsigned short*)alloc((size_t)N * D_INK * 2);
  unsigned short* Wh   = (unsigned short*)alloc((size_t)D_HIDK * D_INK * 2);
  unsigned short* Wl   = (unsigned short*)alloc((size_t)D_HIDK * D_INK * 2);

  int nx = N * D_INK;
  int nw = D_HIDK * D_INK;
  gat0_split_bf16<<<(nx + 255) / 256, 256, 0, stream>>>(x, Xh, Xl, nx);
  gat0_split_bf16<<<(nw + 255) / 256, 256, 0, stream>>>(W0, Wh, Wl, nw);

  int totalTiles = (N >> 4) * 16;
  gat0_gemm_wmma<<<(totalTiles + 7) / 8, 256, 0, stream>>>(Xh, Xl, Wh, Wl, H,
                                                           N);

  int nh = N * HEADSK;
  gat0_attn_dot<<<(nh + 255) / 256, 256, 0, stream>>>(H, as0, ad0, a_s, a_d, N);

  int ntot = N * D_HIDK;
  gat0_init<<<(ntot + 255) / 256, 256, 0, stream>>>(mseg, den, out, N);

  long long eh = (long long)Etot * HEADSK;
  int ehBlocks = (int)((eh + 255) / 256);
  gat0_edge_max<<<ehBlocks, 256, 0, stream>>>(ei, a_s, a_d, mseg, E, Etot);
  gat0_edge_sum<<<ehBlocks, 256, 0, stream>>>(ei, a_s, a_d, mseg, den, E, Etot);

  int aggrBlocks = (int)((eh * 32 + 255) / 256);   // 8 waves / block
  gat0_edge_aggr<<<aggrBlocks, 256, 0, stream>>>(ei, a_s, a_d, mseg, den, H,
                                                 out, E, Etot);

  gat0_elu_bias<<<(ntot + 255) / 256, 256, 0, stream>>>(out, b0, ntot);
}
